// GraphSAGE_48928267436391
// MI455X (gfx1250) — matrix operand order repaired and checked by reading, not compiled, not run on hardware
//
#include <hip/hip_runtime.h>
#include <hip/hip_bf16.h>

typedef __attribute__((ext_vector_type(16))) __bf16 v16bf;
typedef __attribute__((ext_vector_type(8)))  __bf16 v8bf;
typedef __attribute__((ext_vector_type(8)))  float  v8f;

#define NN 50000
#define NE 1600000
#define FD 256      // IN_F == HID == 256 (all aggregation inputs are 256-wide)
#define OD 64

// Guaranteed-hardware fp32 atomic add (no-return form, STOREcnt-tracked).
// Inline asm so we never fall back to a CAS loop for the hot scatter path.
__device__ __forceinline__ void atomicAddF(float* p, float v) {
  asm volatile("global_atomic_add_f32 %0, %1, off"
               :: "v"(p), "v"(v) : "memory");
}

// ---- f32 -> bf16 conversion (x -> activation buffer) ----
__global__ void cvt_f32_bf16(const float* __restrict__ in, __bf16* __restrict__ out, int n) {
  int t = blockIdx.x * blockDim.x + threadIdx.x;
  if (t < n) out[t] = (__bf16)in[t];
}

// ---- transpose+convert weights: W[in][out] f32 -> Wt[out][in] bf16 ----
__global__ void pack_wt(const float* __restrict__ W, __bf16* __restrict__ Wt,
                        int in_d, int out_d) {
  int t = blockIdx.x * blockDim.x + threadIdx.x;
  if (t >= in_d * out_d) return;
  int k = t / out_d, n = t % out_d;
  Wt[(size_t)n * in_d + k] = (__bf16)W[t];
}

// ---- degree (f32 counts via atomic add) ----
__global__ void degree_kernel(const int* __restrict__ dst, float* __restrict__ deg, int ne) {
  int t = blockIdx.x * blockDim.x + threadIdx.x;
  if (t < ne) atomicAddF(&deg[dst[t]], 1.0f);
}

__global__ void inv_deg_kernel(const float* __restrict__ deg, float* __restrict__ invd, int n) {
  int t = blockIdx.x * blockDim.x + threadIdx.x;
  if (t < n) invd[t] = 1.0f / fmaxf(deg[t], 1.0f);
}

// ---- edge scatter: agg[dst] += h[src]; bf16 gather (L2-resident), f32 atomics ----
// 32 threads per edge, 8 features each (one 16B bf16 vector load per thread)
__global__ void scatter_agg(const __bf16* __restrict__ h, const int* __restrict__ src,
                            const int* __restrict__ dst, float* __restrict__ agg, int ne) {
  int t = blockIdx.x * blockDim.x + threadIdx.x;
  int e = t >> 5;
  if (e >= ne) return;
  int c = (t & 31) << 3;
  int s = src[e], d = dst[e];
  v8bf hv = *(const v8bf*)(h + (size_t)s * FD + c);
  float* ap = agg + (size_t)d * FD + c;
#pragma unroll
  for (int i = 0; i < 8; ++i) atomicAddF(ap + i, (float)hv[i]);
}

// ---- fused SAGE layer GEMM: out = relu?(h@Ws + (agg*invd)@Wn + b) ----
// One wave computes a 16x32 output tile (two 16x16 WMMA tiles sharing the A
// fragments); alternating accumulators break the WMMA->WMMA RAW chain.
template <int DO_RELU, bool WRITE_B, bool WRITE_F>
__global__ __launch_bounds__(256)
void sage_gemm_wmma(const __bf16* __restrict__ hin, const float* __restrict__ agg,
                    const float* __restrict__ invd,
                    const __bf16* __restrict__ WtS, const __bf16* __restrict__ WtN,
                    const float* __restrict__ bias,
                    int in_d, int out_d,
                    __bf16* __restrict__ outB, float* __restrict__ outF) {
  const int lane = threadIdx.x & 31;
  const int wave = blockIdx.x * (blockDim.x >> 5) + (threadIdx.x >> 5);
  const int npt  = out_d >> 5;                 // 16x32 tile-pairs per row strip
  const int total = (NN >> 4) * npt;
  if (wave >= total) return;                   // wave-uniform: EXEC all-1s for WMMA
  const int row0 = (wave / npt) << 4;
  const int col0 = (wave % npt) << 5;
  const int g = lane >> 4;                     // lane group (0/1)
  const int m = lane & 15;

  const int arow = row0 + m;                   // A layout: M = lane%16 for both halves
  const float idg = invd[arow];
  const __bf16* hrow = hin + (size_t)arow * in_d;
  const float*  nrow = agg + (size_t)arow * in_d;
  // B layout: lane holds col n, K = 16*g + e (contiguous rows of transposed weights)
  const __bf16* wS0 = WtS + (size_t)(col0 + m) * in_d + 16 * g;
  const __bf16* wS1 = WtS + (size_t)(col0 + 16 + m) * in_d + 16 * g;
  const __bf16* wN0 = WtN + (size_t)(col0 + m) * in_d + 16 * g;
  const __bf16* wN1 = WtN + (size_t)(col0 + 16 + m) * in_d + 16 * g;

  v8f acc0 = {};
  v8f acc1 = {};
  for (int k0 = 0; k0 < in_d; k0 += 32) {
    union { v16bf v; unsigned u[8]; } As, An;
#pragma unroll
    for (int v = 0; v < 8; ++v) {
      // 16-bit A-matrix layout: VGPR v holds K pair at 16g+2v (contiguous, matches B)
      const int c = k0 + 2 * v + 16 * g;
      As.u[v] = *(const unsigned*)(hrow + c);            // 2 packed bf16
      float2 f = *(const float2*)(nrow + c);             // fuse mean-normalize here
      An.v[2 * v]     = (__bf16)(f.x * idg);
      An.v[2 * v + 1] = (__bf16)(f.y * idg);
    }
    v16bf bS0 = *(const v16bf*)(wS0 + k0);               // 32B contiguous
    v16bf bS1 = *(const v16bf*)(wS1 + k0);
    v16bf bN0 = *(const v16bf*)(wN0 + k0);
    v16bf bN1 = *(const v16bf*)(wN1 + k0);
    acc0 = __builtin_amdgcn_wmma_f32_16x16x32_bf16(false, As.v, false, bS0,
                                                   (short)0, acc0, false, false);
    acc1 = __builtin_amdgcn_wmma_f32_16x16x32_bf16(false, As.v, false, bS1,
                                                   (short)0, acc1, false, false);
    acc0 = __builtin_amdgcn_wmma_f32_16x16x32_bf16(false, An.v, false, bN0,
                                                   (short)0, acc0, false, false);
    acc1 = __builtin_amdgcn_wmma_f32_16x16x32_bf16(false, An.v, false, bN1,
                                                   (short)0, acc1, false, false);
  }

  const int  ocol0 = col0 + m;
  const int  ocol1 = col0 + 16 + m;
  const float bv0 = bias[ocol0];
  const float bv1 = bias[ocol1];
#pragma unroll
  for (int r = 0; r < 8; ++r) {                // C layout: M = r + 8g, N = lane%16
    const size_t orow = (size_t)(row0 + r + 8 * g) * out_d;
    float v0 = acc0[r] + bv0;
    float v1 = acc1[r] + bv1;
    if (DO_RELU) { v0 = fmaxf(v0, 0.0f); v1 = fmaxf(v1, 0.0f); }
    if (WRITE_F) { outF[orow + ocol0] = v0; outF[orow + ocol1] = v1; }
    if (WRITE_B) { outB[orow + ocol0] = (__bf16)v0; outB[orow + ocol1] = (__bf16)v1; }
  }
}

extern "C" void kernel_launch(void* const* d_in, const int* in_sizes, int n_in,
                              void* d_out, int out_size, void* d_ws, size_t ws_size,
                              hipStream_t stream) {
  const float* x   = (const float*)d_in[0];
  const int*   src = (const int*)d_in[1];
  const int*   dst = (const int*)d_in[2];
  const float* Ws0 = (const float*)d_in[3];
  const float* Wn0 = (const float*)d_in[4];
  const float* b0  = (const float*)d_in[5];
  const float* Ws1 = (const float*)d_in[6];
  const float* Wn1 = (const float*)d_in[7];
  const float* b1  = (const float*)d_in[8];
  const float* Ws2 = (const float*)d_in[9];
  const float* Wn2 = (const float*)d_in[10];
  const float* b2  = (const float*)d_in[11];
  float* out = (float*)d_out;
  float* out_h  = out;                       // [50000, 64]  (final layer)
  float* out_h1 = out + (size_t)NN * OD;     // [50000, 256] (last_non_exiting)

  char* ws = (char*)d_ws;
  size_t off = 0;
  auto take = [&](size_t bytes) -> void* {
    void* p = ws + off;
    off = (off + bytes + 255) & ~(size_t)255;
    return p;
  };
  float*  agg  = (float*)take((size_t)NN * FD * 4);   // 51.2 MB
  float*  deg  = (float*)take((size_t)NN * 4);
  float*  ivd  = (float*)take((size_t)NN * 4);
  __bf16* hbA  = (__bf16*)take((size_t)NN * FD * 2);  // 25.6 MB
  __bf16* hbB  = (__bf16*)take((size_t)NN * FD * 2);  // 25.6 MB
  __bf16* wtS0 = (__bf16*)take((size_t)FD * FD * 2);
  __bf16* wtN0 = (__bf16*)take((size_t)FD * FD * 2);
  __bf16* wtS1 = (__bf16*)take((size_t)FD * FD * 2);
  __bf16* wtN1 = (__bf16*)take((size_t)FD * FD * 2);
  __bf16* wtS2 = (__bf16*)take((size_t)FD * OD * 2);
  __bf16* wtN2 = (__bf16*)take((size_t)FD * OD * 2);

  // --- degree + inverse-degree (once) ---
  hipMemsetAsync(deg, 0, (size_t)NN * 4, stream);
  degree_kernel<<<(NE + 255) / 256, 256, 0, stream>>>(dst, deg, NE);
  inv_deg_kernel<<<(NN + 255) / 256, 256, 0, stream>>>(deg, ivd, NN);

  // --- convert x to bf16; pack all weights (transposed bf16) ---
  cvt_f32_bf16<<<(NN * FD + 255) / 256, 256, 0, stream>>>(x, hbA, NN * FD);
  pack_wt<<<(FD * FD + 255) / 256, 256, 0, stream>>>(Ws0, wtS0, FD, FD);
  pack_wt<<<(FD * FD + 255) / 256, 256, 0, stream>>>(Wn0, wtN0, FD, FD);
  pack_wt<<<(FD * FD + 255) / 256, 256, 0, stream>>>(Ws1, wtS1, FD, FD);
  pack_wt<<<(FD * FD + 255) / 256, 256, 0, stream>>>(Wn1, wtN1, FD, FD);
  pack_wt<<<(FD * OD + 255) / 256, 256, 0, stream>>>(Ws2, wtS2, FD, OD);
  pack_wt<<<(FD * OD + 255) / 256, 256, 0, stream>>>(Wn2, wtN2, FD, OD);

  const int scat_blocks = (NE * 32 + 255) / 256;
  const int pairs256 = (NN / 16) * (FD / 32);       // 25000 wave-tiles
  const int pairs64  = (NN / 16) * (OD / 32);       // 6250 wave-tiles
  const int gemm_blocks256 = (pairs256 + 7) / 8;    // 8 waves per block
  const int gemm_blocks64  = (pairs64 + 7) / 8;

  // --- layer 0: relu(x@Ws0 + mean@Wn0 + b0) -> hbB (bf16) ---
  hipMemsetAsync(agg, 0, (size_t)NN * FD * 4, stream);
  scatter_agg<<<scat_blocks, 256, 0, stream>>>(hbA, src, dst, agg, NE);
  sage_gemm_wmma<1, true, false><<<gemm_blocks256, 256, 0, stream>>>(
      hbA, agg, ivd, wtS0, wtN0, b0, FD, FD, hbB, nullptr);

  // --- layer 1: relu(...) -> hbA (bf16) and out_h1 (f32) ---
  hipMemsetAsync(agg, 0, (size_t)NN * FD * 4, stream);
  scatter_agg<<<scat_blocks, 256, 0, stream>>>(hbB, src, dst, agg, NE);
  sage_gemm_wmma<1, true, true><<<gemm_blocks256, 256, 0, stream>>>(
      hbB, agg, ivd, wtS1, wtN1, b1, FD, FD, hbA, out_h1);

  // --- layer 2: (no relu) -> out_h (f32) ---
  hipMemsetAsync(agg, 0, (size_t)NN * FD * 4, stream);
  scatter_agg<<<scat_blocks, 256, 0, stream>>>(hbA, src, dst, agg, NE);
  sage_gemm_wmma<0, false, true><<<gemm_blocks64, 256, 0, stream>>>(
      hbA, agg, ivd, wtS2, wtN2, b2, FD, OD, nullptr, out_h);
}
